// ElementwiseTensorProduct_2911987826947
// MI455X (gfx1250) — compile-verified, matrix operand order checked by reading, not run on hardware
//
#include <hip/hip_runtime.h>
#include <hip/hip_bf16.h>

// out[z,k] = sum_{i,j} W[k,i,j] * f1[z,i] * f2[z,j]
// Z=16384, I=J=128, K=192.
//
// GEMM form: out = P @ Wr^T with P[z, ij] = f1[z,i]*f2[z,j] synthesized
// on the fly in registers (f2 A-fragment scaled by f1 scalar), Wr^T
// pre-packed in bf16 into d_ws in the exact CDNA5 16-bit B(32x16) lane
// layout so each fragment is one contiguous 32B load per lane, and the
// three fragments a wave needs per step sit at constant immediate offsets
// from a single per-wave pointer (one u64 add per inner iteration).
//
// d_ws requirement: 192*128*128 bf16 = 6,291,456 bytes.

#define Z_TOT   16384
#define DIM_IN  128
#define DIM_OUT 192

typedef __attribute__((ext_vector_type(16))) __bf16 v16bf;
typedef __attribute__((ext_vector_type(8)))  float  v8f;

// ---------------------------------------------------------------------------
// Pre-pack: W (f32, [k][i][j]) -> bf16 B-fragments.
//   idx = ((c*12 + kt)*32 + lane)*16 + e
//   c  = ij-chunk (i = c>>2, jc = c&3), 512 chunks; kt = k-tile (12)
//   lane 0-15  -> N = lane,    K = 0..7, 16..23   (j offsets 2v+p / 2v+8+p)
//   lane 16-31 -> N = lane-16, K = 8..15, 24..31
// ---------------------------------------------------------------------------
__global__ __launch_bounds__(256)
void etp_wpack_kernel(const float* __restrict__ W, __hip_bfloat16* __restrict__ Wb) {
    int idx = blockIdx.x * 256 + threadIdx.x;           // 0 .. 3145727
    int e   = idx & 15;
    int l   = (idx >> 4) & 31;
    int ckt = idx >> 9;
    int kt  = ckt % 12;
    int c   = ckt / 12;

    int p  = e & 1;
    int v  = e >> 1;
    int i  = c >> 2;
    int jc = c & 3;
    int k  = kt * 16 + (l & 15);
    int hi = l >> 4;
    int koff = (v < 4 ? 2 * v : 2 * v + 8) + hi * 8 + p;   // K index 0..31
    int j  = jc * 32 + koff;

    float val = W[(k * DIM_IN + i) * DIM_IN + j];
    Wb[idx] = __float2bfloat16(val);
}

// ---------------------------------------------------------------------------
// Main kernel: workgroup = 64 z-rows x all 192 k, 8 waves (256 threads).
//   wave -> (z-half = wave>>2, k-quarter kq = wave&3)
//   each wave: 3 k-tiles (kt = kq + 4*kk) x 2 z-subtiles = 6 accumulators
//   inner iteration: 16 v_pk_mul_bf16 (A synth) + 6 b128 loads (constant
//   immediate offsets off one pointer) + 6 WMMAs.
// ---------------------------------------------------------------------------
__global__ __launch_bounds__(256)
void etp_main_kernel(const float* __restrict__ f1,
                     const float* __restrict__ f2,
                     const __hip_bfloat16* __restrict__ Wb,
                     float* __restrict__ out) {
    __shared__ float lf1[DIM_IN * 64];   // [i][z] transposed, conflict-free reads
    __shared__ float lf2[64 * DIM_IN];   // [z][j]

    const int tid   = threadIdx.x;
    const int lane  = tid & 31;
    const int wave  = tid >> 5;
    const int zhalf = wave >> 2;         // 0/1 -> z rows 0..31 / 32..63
    const int kq    = wave & 3;          // k-tile residue class mod 4
    const int z0    = blockIdx.x * 64;

    // Stage tiles into LDS (8192 floats each, 32 per thread).
    for (int t = tid; t < 64 * DIM_IN; t += 256) {
        int z = t >> 7;
        int i = t & 127;
        lf1[i * 64 + z] = f1[(size_t)(z0 + z) * DIM_IN + i];
        lf2[t]          = f2[(size_t)(z0 + z) * DIM_IN + i];
    }
    __syncthreads();

    const int ln16 = lane & 15;
    const int hi   = lane >> 4;
    const int zb   = zhalf * 32;         // wave's z base within block

    v8f acc[3][2];
    #pragma unroll
    for (int kk = 0; kk < 3; ++kk)
        #pragma unroll
        for (int zs = 0; zs < 2; ++zs)
            acc[kk][zs] = (v8f){0.f, 0.f, 0.f, 0.f, 0.f, 0.f, 0.f, 0.f};

    for (int jc = 0; jc < 4; ++jc) {
        // f2 A-fragment skeleton for this j-chunk, per z-subtile
        // (16-bit A 16x32 layout: row M = lane%16; lane-half picks K 0-7,16-23 / 8-15,24-31).
        v16bf fr2[2];
        #pragma unroll
        for (int zs = 0; zs < 2; ++zs) {
            int z = zb + zs * 16 + ln16;
            #pragma unroll
            for (int v = 0; v < 8; ++v) {
                int j = jc * 32 + (v < 4 ? 2 * v : 2 * v + 8) + hi * 8;
                fr2[zs][2 * v]     = (__bf16)lf2[z * DIM_IN + j];
                fr2[zs][2 * v + 1] = (__bf16)lf2[z * DIM_IN + j + 1];
            }
        }

        // One pointer per wave; three fragments live at +0 / +4096B / +8192B.
        // c = i*4 + jc; fragment base elem = (c*12 + kq)*512 + lane*16.
        const __hip_bfloat16* wp =
            Wb + ((size_t)(jc * 12 + kq) * 32 + lane) * 16;

        for (int i = 0; i < DIM_IN; ++i) {
            // Per-row scalar f1[z, i]; scale the f2 fragment to synthesize P.
            __bf16 s0 = (__bf16)lf1[i * 64 + zb + ln16];
            __bf16 s1 = (__bf16)lf1[i * 64 + zb + 16 + ln16];
            v16bf a0, a1;
            #pragma unroll
            for (int e = 0; e < 16; ++e) {
                a0[e] = s0 * fr2[0][e];
                a1[e] = s1 * fr2[1][e];
            }

            // kt = kq + 4*kk  ->  +kk * (4*32*16) elements = +kk*4096 bytes
            const v16bf B0 = *(const v16bf*)(const void*)(wp);
            const v16bf B1 = *(const v16bf*)(const void*)(wp + 2048);
            const v16bf B2 = *(const v16bf*)(const void*)(wp + 4096);

            acc[0][0] = __builtin_amdgcn_wmma_f32_16x16x32_bf16(
                false, a0, false, B0, (short)0, acc[0][0], false, false);
            acc[0][1] = __builtin_amdgcn_wmma_f32_16x16x32_bf16(
                false, a1, false, B0, (short)0, acc[0][1], false, false);
            acc[1][0] = __builtin_amdgcn_wmma_f32_16x16x32_bf16(
                false, a0, false, B1, (short)0, acc[1][0], false, false);
            acc[1][1] = __builtin_amdgcn_wmma_f32_16x16x32_bf16(
                false, a1, false, B1, (short)0, acc[1][1], false, false);
            acc[2][0] = __builtin_amdgcn_wmma_f32_16x16x32_bf16(
                false, a0, false, B2, (short)0, acc[2][0], false, false);
            acc[2][1] = __builtin_amdgcn_wmma_f32_16x16x32_bf16(
                false, a1, false, B2, (short)0, acc[2][1], false, false);

            wp += 4 * 12 * 32 * 16;   // c += 4 -> +24576 elems (+49152 B)
        }
    }

    // C fragment layout: VGPR r, lane -> (M = r + 8*hi, N = lane%16).
    #pragma unroll
    for (int kk = 0; kk < 3; ++kk) {
        const int kt = kq + kk * 4;
        #pragma unroll
        for (int zs = 0; zs < 2; ++zs) {
            #pragma unroll
            for (int r = 0; r < 8; ++r) {
                int z = z0 + zb + zs * 16 + r + 8 * hi;
                out[(size_t)z * DIM_OUT + kt * 16 + ln16] = acc[kk][zs][r];
            }
        }
    }
}

extern "C" void kernel_launch(void* const* d_in, const int* in_sizes, int n_in,
                              void* d_out, int out_size, void* d_ws, size_t ws_size,
                              hipStream_t stream) {
    const float* f1 = (const float*)d_in[0];   // (Z, 128) f32
    const float* f2 = (const float*)d_in[1];   // (Z, 128) f32
    const float* W  = (const float*)d_in[2];   // (192, 128, 128) f32
    float* out = (float*)d_out;                // (Z, 192) f32
    __hip_bfloat16* Wb = (__hip_bfloat16*)d_ws; // needs 6,291,456 bytes

    const int wb_elems = DIM_OUT * DIM_IN * DIM_IN;            // 3,145,728
    etp_wpack_kernel<<<wb_elems / 256, 256, 0, stream>>>(W, Wb);

    etp_main_kernel<<<Z_TOT / 64, 256, 0, stream>>>(f1, f2, Wb, out);
}
